// Attention_61229053772066
// MI455X (gfx1250) — matrix operand order repaired and checked by reading, not compiled, not run on hardware
//
#include <hip/hip_runtime.h>
#include <cstdint>

typedef unsigned short u16;
typedef unsigned int   u32;
typedef __attribute__((ext_vector_type(16))) __bf16 v16bf;
typedef __attribute__((ext_vector_type(8)))  float  v8f;
typedef __attribute__((ext_vector_type(4)))  u32    u32x4;

#define AS1 __attribute__((address_space(1)))

union Frag {
    v16bf v;
    u32x4 u[2];
};

__device__ __forceinline__ u16 f2bf(float f) {
    u32 u = __float_as_uint(f);
    u32 r = u + 0x7FFFu + ((u >> 16) & 1u);   // round-to-nearest-even
    return (u16)(r >> 16);
}
__device__ __forceinline__ float bf2f(u16 h) {
    return __uint_as_float(((u32)h) << 16);
}

// Explicit global-address-space accessors: force global_load_b128 /
// global_store (LOADcnt/STOREcnt only), never flat_*.
__device__ __forceinline__ u32x4 gload4(const u16* p) {
    return *(const AS1 u32x4*)p;
}
__device__ __forceinline__ void gstore16(u16* p, u16 v) {
    *(AS1 u16*)p = v;
}
__device__ __forceinline__ void gstore32(float* p, float v) {
    *(AS1 float*)p = v;
}

#define WMMA_BF16(Afrag, Bfrag, Cacc) \
    __builtin_amdgcn_wmma_f32_16x16x32_bf16(false, (Afrag), false, (Bfrag), (short)0, (Cacc), false, false)

// ---------------------------------------------------------------------------
// fp32 -> bf16 elementwise convert (x)
// ---------------------------------------------------------------------------
__global__ void cvt_bf16_kernel(const float* __restrict__ in, u16* __restrict__ out, int n4) {
    int i = blockIdx.x * blockDim.x + threadIdx.x;
    if (i >= n4) return;
    float4 f = ((const float4*)in)[i];
    ushort4 o;
    o.x = f2bf(f.x); o.y = f2bf(f.y); o.z = f2bf(f.z); o.w = f2bf(f.w);
    ((ushort4*)out)[i] = o;
}

// ---------------------------------------------------------------------------
// fp32 (K x N) -> bf16 transposed (N x K), LDS-tiled
// ---------------------------------------------------------------------------
__global__ void cvt_transpose_kernel(const float* __restrict__ W, u16* __restrict__ Wt,
                                     int K, int N) {
    __shared__ float tile[32][33];
    int kb = blockIdx.y * 32, nb = blockIdx.x * 32;
    int tx = threadIdx.x, ty = threadIdx.y;
    for (int j = ty; j < 32; j += 8)
        tile[j][tx] = W[(size_t)(kb + j) * N + nb + tx];
    __syncthreads();
    for (int j = ty; j < 32; j += 8)
        Wt[(size_t)(nb + j) * K + kb + tx] = f2bf(tile[tx][j]);
}

// ---------------------------------------------------------------------------
// bf16 (M x 128) -> bf16 vT (128 x M per batch), for attn @ V fragments
// ---------------------------------------------------------------------------
__global__ void transpose_v_kernel(const u16* __restrict__ v, u16* __restrict__ vT, int T) {
    __shared__ u16 tile[32][33];
    int b = blockIdx.z;
    int tb = blockIdx.x * 32;   // t tile
    int db = blockIdx.y * 32;   // d tile
    int tx = threadIdx.x, ty = threadIdx.y;
    for (int j = ty; j < 32; j += 8)
        tile[j][tx] = v[(size_t)(b * T + tb + j) * 128 + db + tx];
    __syncthreads();
    for (int j = ty; j < 32; j += 8)
        vT[(size_t)(b * 128 + db + j) * T + tb + tx] = tile[tx][j];
}

// ---------------------------------------------------------------------------
// In-place RoPE on bf16 tensor shaped (B*T, H, 128); H=16 for q, H=1 for k.
// ---------------------------------------------------------------------------
__global__ void rope_kernel(u16* __restrict__ t, const float* __restrict__ cosT,
                            const float* __restrict__ sinT, int total, int T, int H) {
    int idx = blockIdx.x * blockDim.x + threadIdx.x;
    if (idx >= total) return;
    int i    = idx & 63;        // pair index 0..63
    int rest = idx >> 6;
    int h    = rest % H;
    int bt   = rest / H;
    int tt   = bt % T;
    size_t base = ((size_t)bt * H + h) * 128 + 2 * i;
    u32 pr = *(const AS1 u32*)(t + base);
    float e = bf2f((u16)(pr & 0xFFFFu));
    float o = bf2f((u16)(pr >> 16));
    float c = cosT[tt * 64 + i], s = sinT[tt * 64 + i];
    float re = e * c - o * s;
    float ro = e * s + o * c;
    *(AS1 u32*)(t + base) = (u32)f2bf(re) | ((u32)f2bf(ro) << 16);
}

// ---------------------------------------------------------------------------
// Tiled WMMA GEMM: C[M,N] = A[M,K] * Bt[N,K]^T  (bf16 in, fp32 accum)
// BM=128 BN=128 BK=64, 256 threads = 8 waves, wave w owns rows [w*16, w*16+16)
// Double-buffered LDS; next tile's global loads overlap the current WMMAs.
// ---------------------------------------------------------------------------
template <bool OUT_F32>
__global__ void __launch_bounds__(256) gemm_bf16(const u16* __restrict__ A,
                                                 const u16* __restrict__ Bt,
                                                 void* __restrict__ Cout,
                                                 int M, int N, int K) {
    __shared__ __align__(16) u16 sA[2][128][72];
    __shared__ __align__(16) u16 sB[2][128][72];
    const int m0 = blockIdx.y * 128, n0 = blockIdx.x * 128;
    const int tid  = threadIdx.x;
    const int lane = tid & 31, w = tid >> 5;
    const int hf = lane >> 4, ln = lane & 15;
    const int lrow = tid >> 1;            // 0..127
    const int lcol = (tid & 1) * 32;      // 0 or 32

    const u16* ga = A  + (size_t)(m0 + lrow) * K + lcol;
    const u16* gb = Bt + (size_t)(n0 + lrow) * K + lcol;

    v8f acc[8];
#pragma unroll
    for (int nt = 0; nt < 8; ++nt)
#pragma unroll
        for (int r = 0; r < 8; ++r) acc[nt][r] = 0.0f;

    u32x4 ra[4], rb[4];
    // preload tile 0 straight into buffer 0
#pragma unroll
    for (int j = 0; j < 4; ++j) {
        ra[j] = gload4(ga + 8 * j);
        rb[j] = gload4(gb + 8 * j);
    }
#pragma unroll
    for (int j = 0; j < 4; ++j) {
        *(u32x4*)&sA[0][lrow][lcol + 8 * j] = ra[j];
        *(u32x4*)&sB[0][lrow][lcol + 8 * j] = rb[j];
    }
    __syncthreads();

    const int nk = K >> 6;   // K / 64
    int cur = 0;
    for (int kt = 0; kt < nk; ++kt) {
        const bool more = (kt + 1 < nk);
        if (more) {
            const u16* ga2 = ga + (kt + 1) * 64;
            const u16* gb2 = gb + (kt + 1) * 64;
            if (kt + 2 < nk) {                       // -> global_prefetch_b8
                __builtin_prefetch(ga2 + 64, 0, 1);
                __builtin_prefetch(gb2 + 64, 0, 1);
            }
#pragma unroll
            for (int j = 0; j < 4; ++j) {
                ra[j] = gload4(ga2 + 8 * j);
                rb[j] = gload4(gb2 + 8 * j);
            }
        }
        // ---- compute on buffer `cur` (hides the global loads above)
#pragma unroll
        for (int ks = 0; ks < 2; ++ks) {
            Frag fa;
            const u16* pa = &sA[cur][w * 16 + ln][ks * 32 + hf * 16];
            fa.u[0] = *(const u32x4*)(pa);
            fa.u[1] = *(const u32x4*)(pa + 8);
#pragma unroll
            for (int g = 0; g < 2; ++g) {            // n-tiles in groups of 4
                Frag f0, f1, f2, f3;
                const u16* pb0 = &sB[cur][(g * 4 + 0) * 16 + ln][ks * 32 + hf * 16];
                const u16* pb1 = &sB[cur][(g * 4 + 1) * 16 + ln][ks * 32 + hf * 16];
                const u16* pb2 = &sB[cur][(g * 4 + 2) * 16 + ln][ks * 32 + hf * 16];
                const u16* pb3 = &sB[cur][(g * 4 + 3) * 16 + ln][ks * 32 + hf * 16];
                f0.u[0] = *(const u32x4*)(pb0); f0.u[1] = *(const u32x4*)(pb0 + 8);
                f1.u[0] = *(const u32x4*)(pb1); f1.u[1] = *(const u32x4*)(pb1 + 8);
                f2.u[0] = *(const u32x4*)(pb2); f2.u[1] = *(const u32x4*)(pb2 + 8);
                f3.u[0] = *(const u32x4*)(pb3); f3.u[1] = *(const u32x4*)(pb3 + 8);
                acc[g * 4 + 0] = WMMA_BF16(fa.v, f0.v, acc[g * 4 + 0]);
                acc[g * 4 + 1] = WMMA_BF16(fa.v, f1.v, acc[g * 4 + 1]);
                acc[g * 4 + 2] = WMMA_BF16(fa.v, f2.v, acc[g * 4 + 2]);
                acc[g * 4 + 3] = WMMA_BF16(fa.v, f3.v, acc[g * 4 + 3]);
            }
        }
        // ---- stage next tile into the other buffer
        if (more) {
#pragma unroll
            for (int j = 0; j < 4; ++j) {
                *(u32x4*)&sA[cur ^ 1][lrow][lcol + 8 * j] = ra[j];
                *(u32x4*)&sB[cur ^ 1][lrow][lcol + 8 * j] = rb[j];
            }
        }
        __syncthreads();
        cur ^= 1;
    }

#pragma unroll
    for (int nt = 0; nt < 8; ++nt)
#pragma unroll
        for (int r = 0; r < 8; ++r) {
            int m = m0 + w * 16 + hf * 8 + r;
            int n = n0 + nt * 16 + ln;
            if (OUT_F32)
                gstore32((float*)Cout + (size_t)m * N + n, acc[nt][r]);
            else
                gstore16((u16*)Cout + (size_t)m * N + n, f2bf(acc[nt][r]));
        }
}

// ---------------------------------------------------------------------------
// Flash attention (MQA): grid(T/64, H, B), 128 threads = 4 waves.
// k/v key-blocks cooperatively staged in double-buffered LDS (global traffic
// /4, latency hidden behind WMMA phases). Wave w owns 16 query rows.
// ---------------------------------------------------------------------------
__global__ void __launch_bounds__(128) flash_kernel(const u16* __restrict__ q,
                                                    const u16* __restrict__ k,
                                                    const u16* __restrict__ vT,
                                                    u16* __restrict__ y,
                                                    int T, float scale) {
    __shared__ __align__(16) u16 sK[2][64][136];   // key-block: 64 keys x 128 d
    __shared__ __align__(16) u16 sV[2][128][72];   // v-block:  128 d x 64 keys
    __shared__ __align__(16) u16 sP[4][16][72];    // per-wave P staging
    const int H = 16, D = 128;
    const int b = blockIdx.z, h = blockIdx.y, qblk = blockIdx.x;
    const int q0 = qblk * 64;
    const int tid = threadIdx.x;
    const int lane = tid & 31, w = tid >> 5;
    const int hf = lane >> 4, ln = lane & 15;
    // cooperative tile-load coordinates
    const int krow_l = tid >> 1;            // 0..63
    const int kcol_l = (tid & 1) * 64;      // 0 or 64
    const int vrow_l = tid;                 // 0..127

    const u16* kgbase = k + ((size_t)b * T) * D;
    const u16* vgbase = vT + ((size_t)b * 128 + vrow_l) * T;

    // Preload q A-fragments (row = lane&15, contiguous k = 16*(lane>>4) + elem)
    Frag qa[4];
    const u16* qrow = q + ((size_t)(b * T + q0 + w * 16 + ln)) * (H * D) + h * D;
#pragma unroll
    for (int ks = 0; ks < 4; ++ks) {
        qa[ks].u[0] = gload4(qrow + ks * 32 + hf * 16);
        qa[ks].u[1] = gload4(qrow + ks * 32 + 8 + hf * 16);
    }

    float mrow[8], lsum[8];
    v8f O[8];
#pragma unroll
    for (int r = 0; r < 8; ++r) { mrow[r] = -1e38f; lsum[r] = 0.0f; }
#pragma unroll
    for (int nt = 0; nt < 8; ++nt)
#pragma unroll
        for (int r = 0; r < 8; ++r) O[nt][r] = 0.0f;

    // ---- preload key-block 0 into buffer 0
    u32x4 rk[8], rv[8];
    {
        const u16* gk = kgbase + (size_t)krow_l * D + kcol_l;
        const u16* gv = vgbase;
#pragma unroll
        for (int j = 0; j < 8; ++j) rk[j] = gload4(gk + 8 * j);
#pragma unroll
        for (int j = 0; j < 8; ++j) rv[j] = gload4(gv + 8 * j);
#pragma unroll
        for (int j = 0; j < 8; ++j) *(u32x4*)&sK[0][krow_l][kcol_l + 8 * j] = rk[j];
#pragma unroll
        for (int j = 0; j < 8; ++j) *(u32x4*)&sV[0][vrow_l][8 * j] = rv[j];
    }
    __syncthreads();

    int cur = 0;
    for (int kb = 0; kb <= qblk; ++kb) {
        const bool more = (kb < qblk);
        // ---- issue next k-tile global loads (consumed after S compute)
        if (more) {
            const u16* gk = kgbase + (size_t)((kb + 1) * 64 + krow_l) * D + kcol_l;
            if (kb + 2 <= qblk)                      // -> global_prefetch_b8
                __builtin_prefetch(gk + 64 * D, 0, 1);
#pragma unroll
            for (int j = 0; j < 8; ++j) rk[j] = gload4(gk + 8 * j);
        }
        // ---- scores S = q @ k^T (16 x 64) from sK[cur]
        v8f S[4];
#pragma unroll
        for (int nt = 0; nt < 4; ++nt)
#pragma unroll
            for (int r = 0; r < 8; ++r) S[nt][r] = 0.0f;
#pragma unroll
        for (int ks = 0; ks < 4; ++ks) {
            Frag f0, f1, f2, f3;
            const u16* p0 = &sK[cur][0 * 16 + ln][ks * 32 + hf * 16];
            const u16* p1 = &sK[cur][1 * 16 + ln][ks * 32 + hf * 16];
            const u16* p2 = &sK[cur][2 * 16 + ln][ks * 32 + hf * 16];
            const u16* p3 = &sK[cur][3 * 16 + ln][ks * 32 + hf * 16];
            f0.u[0] = *(const u32x4*)(p0); f0.u[1] = *(const u32x4*)(p0 + 8);
            f1.u[0] = *(const u32x4*)(p1); f1.u[1] = *(const u32x4*)(p1 + 8);
            f2.u[0] = *(const u32x4*)(p2); f2.u[1] = *(const u32x4*)(p2 + 8);
            f3.u[0] = *(const u32x4*)(p3); f3.u[1] = *(const u32x4*)(p3 + 8);
            S[0] = WMMA_BF16(qa[ks].v, f0.v, S[0]);
            S[1] = WMMA_BF16(qa[ks].v, f1.v, S[1]);
            S[2] = WMMA_BF16(qa[ks].v, f2.v, S[2]);
            S[3] = WMMA_BF16(qa[ks].v, f3.v, S[3]);
        }
        // ---- scale + causal mask (diagonal block only)
        const bool diag = (kb == qblk);
#pragma unroll
        for (int nt = 0; nt < 4; ++nt)
#pragma unroll
            for (int r = 0; r < 8; ++r) {
                float sv = S[nt][r] * scale;
                if (diag) {
                    int key = kb * 64 + nt * 16 + ln;
                    int row = q0 + w * 16 + hf * 8 + r;
                    if (key > row) sv = -1e30f;
                }
                S[nt][r] = sv;
            }
        // ---- online softmax (row stats via shfl_xor within each 16-lane half)
#pragma unroll
        for (int r = 0; r < 8; ++r) {
            float m = fmaxf(fmaxf(S[0][r], S[1][r]), fmaxf(S[2][r], S[3][r]));
#pragma unroll
            for (int msk = 1; msk < 16; msk <<= 1)
                m = fmaxf(m, __shfl_xor(m, msk, 32));
            float mnew  = fmaxf(mrow[r], m);
            float alpha = __expf(mrow[r] - mnew);
            mrow[r] = mnew;
            float ssum = 0.0f;
#pragma unroll
            for (int nt = 0; nt < 4; ++nt) {
                float p = __expf(S[nt][r] - mnew);
                S[nt][r] = p;
                ssum += p;
            }
#pragma unroll
            for (int msk = 1; msk < 16; msk <<= 1)
                ssum += __shfl_xor(ssum, msk, 32);
            lsum[r] = lsum[r] * alpha + ssum;
#pragma unroll
            for (int nt = 0; nt < 8; ++nt) O[nt][r] *= alpha;
        }
        // ---- store next k-tile; issue next v-tile loads (consumed after PV)
        if (more) {
#pragma unroll
            for (int j = 0; j < 8; ++j)
                *(u32x4*)&sK[cur ^ 1][krow_l][kcol_l + 8 * j] = rk[j];
            const u16* gv = vgbase + (kb + 1) * 64;
#pragma unroll
            for (int j = 0; j < 8; ++j) rv[j] = gload4(gv + 8 * j);
        }
        // ---- stage P to LDS (C-layout -> A-layout transpose, per-wave slice)
        __syncthreads();
#pragma unroll
        for (int nt = 0; nt < 4; ++nt)
#pragma unroll
            for (int r = 0; r < 8; ++r)
                sP[w][hf * 8 + r][nt * 16 + ln] = f2bf(S[nt][r]);
        __syncthreads();
        Frag pa[2];
#pragma unroll
        for (int ks = 0; ks < 2; ++ks) {
            const u16* pp = &sP[w][ln][ks * 32 + hf * 16];
            pa[ks].u[0] = *(const u32x4*)(pp);
            pa[ks].u[1] = *(const u32x4*)(pp + 8);
        }
        // ---- O += P @ V from sV[cur]
#pragma unroll
        for (int ks = 0; ks < 2; ++ks) {
#pragma unroll
            for (int g = 0; g < 2; ++g) {            // d-tiles in groups of 4
                Frag f0, f1, f2, f3;
                const u16* p0 = &sV[cur][(g * 4 + 0) * 16 + ln][ks * 32 + hf * 16];
                const u16* p1 = &sV[cur][(g * 4 + 1) * 16 + ln][ks * 32 + hf * 16];
                const u16* p2 = &sV[cur][(g * 4 + 2) * 16 + ln][ks * 32 + hf * 16];
                const u16* p3 = &sV[cur][(g * 4 + 3) * 16 + ln][ks * 32 + hf * 16];
                f0.u[0] = *(const u32x4*)(p0); f0.u[1] = *(const u32x4*)(p0 + 8);
                f1.u[0] = *(const u32x4*)(p1); f1.u[1] = *(const u32x4*)(p1 + 8);
                f2.u[0] = *(const u32x4*)(p2); f2.u[1] = *(const u32x4*)(p2 + 8);
                f3.u[0] = *(const u32x4*)(p3); f3.u[1] = *(const u32x4*)(p3 + 8);
                O[g * 4 + 0] = WMMA_BF16(pa[ks].v, f0.v, O[g * 4 + 0]);
                O[g * 4 + 1] = WMMA_BF16(pa[ks].v, f1.v, O[g * 4 + 1]);
                O[g * 4 + 2] = WMMA_BF16(pa[ks].v, f2.v, O[g * 4 + 2]);
                O[g * 4 + 3] = WMMA_BF16(pa[ks].v, f3.v, O[g * 4 + 3]);
            }
        }
        // ---- store next v-tile into the back buffer
        if (more) {
#pragma unroll
            for (int j = 0; j < 8; ++j)
                *(u32x4*)&sV[cur ^ 1][vrow_l][8 * j] = rv[j];
        }
        __syncthreads();
        cur ^= 1;
    }
    // ---- normalize and store y (bf16, (B,T,H*D) layout)
#pragma unroll
    for (int r = 0; r < 8; ++r) {
        float inv = 1.0f / lsum[r];
        int row = q0 + w * 16 + hf * 8 + r;
        u16* yrow = y + ((size_t)(b * T + row)) * (H * D) + h * D;
#pragma unroll
        for (int nt = 0; nt < 8; ++nt)
            gstore16(yrow + nt * 16 + ln, f2bf(O[nt][r] * inv));
    }
}

// ---------------------------------------------------------------------------
extern "C" void kernel_launch(void* const* d_in, const int* in_sizes, int n_in,
                              void* d_out, int out_size, void* d_ws, size_t ws_size,
                              hipStream_t stream) {
    (void)in_sizes; (void)n_in; (void)out_size; (void)ws_size;
    const float* x    = (const float*)d_in[0];
    const float* fcos = (const float*)d_in[1];
    const float* fsin = (const float*)d_in[2];
    const float* Wq   = (const float*)d_in[3];
    const float* Wk   = (const float*)d_in[4];
    const float* Wv   = (const float*)d_in[5];
    const float* Wo   = (const float*)d_in[6];
    float* out = (float*)d_out;

    const int B = 2, T = 2048, C = 2048, H = 16, D = 128;
    const int M = B * T;  // 4096

    char* ws = (char*)d_ws;
    size_t off = 0;
    auto alloc = [&](size_t bytes) {
        char* p = ws + off;
        off += (bytes + 255) & ~(size_t)255;
        return p;
    };
    u16* xb  = (u16*)alloc((size_t)M * C * 2);
    u16* Wqt = (u16*)alloc((size_t)C * C * 2);
    u16* Wkt = (u16*)alloc((size_t)C * D * 2);
    u16* Wvt = (u16*)alloc((size_t)C * D * 2);
    u16* Wot = (u16*)alloc((size_t)C * C * 2);
    u16* qb  = (u16*)alloc((size_t)M * C * 2);
    u16* kbf = (u16*)alloc((size_t)M * D * 2);
    u16* vbf = (u16*)alloc((size_t)M * D * 2);
    u16* vT  = (u16*)alloc((size_t)B * D * T * 2);
    u16* yb  = (u16*)alloc((size_t)M * C * 2);

    // 1) converts (x plain; weights converted + transposed to (N x K))
    {
        int n4 = M * C / 4;
        cvt_bf16_kernel<<<(n4 + 255) / 256, 256, 0, stream>>>(x, xb, n4);
    }
    dim3 tb(32, 8);
    cvt_transpose_kernel<<<dim3(C / 32, C / 32), tb, 0, stream>>>(Wq, Wqt, C, C);
    cvt_transpose_kernel<<<dim3(D / 32, C / 32), tb, 0, stream>>>(Wk, Wkt, C, D);
    cvt_transpose_kernel<<<dim3(D / 32, C / 32), tb, 0, stream>>>(Wv, Wvt, C, D);
    cvt_transpose_kernel<<<dim3(C / 32, C / 32), tb, 0, stream>>>(Wo, Wot, C, C);

    // 2) projections
    gemm_bf16<false><<<dim3(C / 128, M / 128), 256, 0, stream>>>(xb, Wqt, (void*)qb, M, C, C);
    gemm_bf16<false><<<dim3(1,       M / 128), 256, 0, stream>>>(xb, Wkt, (void*)kbf, M, D, C);
    gemm_bf16<false><<<dim3(1,       M / 128), 256, 0, stream>>>(xb, Wvt, (void*)vbf, M, D, C);

    // 3) RoPE (in place)
    {
        int totq = B * T * H * (D / 2);
        rope_kernel<<<(totq + 255) / 256, 256, 0, stream>>>(qb, fcos, fsin, totq, T, H);
        int totk = B * T * (D / 2);
        rope_kernel<<<(totk + 255) / 256, 256, 0, stream>>>(kbf, fcos, fsin, totk, T, 1);
    }

    // 4) v transpose for attn@V fragment reads
    transpose_v_kernel<<<dim3(T / 32, D / 32, B), tb, 0, stream>>>(vbf, vT, T);

    // 5) flash attention
    flash_kernel<<<dim3(T / 64, H, B), 128, 0, stream>>>(qb, kbf, vT, yb, T,
                                                         0.08838834764831845f);

    // 6) output projection (fp32 out)
    gemm_bf16<true><<<dim3(C / 128, M / 128), 256, 0, stream>>>(yb, Wot, (void*)out, M, C, C);
}
